// MCRFGNN_29454885716097
// MI455X (gfx1250) — matrix operand order repaired and checked by reading, not compiled, not run on hardware
//
#include <hip/hip_runtime.h>
#include <cstdint>
#include <cstddef>

// ---------------------------------------------------------------------------
// MCRF-GNN pipeline for MI455X (gfx1250, wave32).
// A converted to bf16 once (67MB, L2-resident); all N x N matmuls via
// v_wmma_f32_16x16x32_bf16 with fp32 accumulation. Each wave owns a 2-row-tile
// (BP) or 2x2-tile (GCN) block so B fragments are loaded once per k-step and
// tile loops have uniform trip counts (EXEC stays all-1s around WMMA).
// ---------------------------------------------------------------------------

typedef __bf16 bf16;
typedef bf16  v8bf  __attribute__((ext_vector_type(8)));
typedef bf16  v16bf __attribute__((ext_vector_type(16)));
typedef float v8f   __attribute__((ext_vector_type(8)));

#define BB 128
#define NN 512
#define WW 5
#define SSUP 256
#define EPSF 1e-6f
#define BPI 7

__device__ __forceinline__ bf16 f2bf(float f) {
  unsigned u = __float_as_uint(f);
  unsigned short h = (unsigned short)((u + 0x7FFFu + ((u >> 16) & 1u)) >> 16);
  return __builtin_bit_cast(bf16, h);
}

__device__ __forceinline__ float wave_sum32(float v) {
  for (int o = 16; o > 0; o >>= 1) v += __shfl_down(v, o, 32);
  return v;
}

__device__ __forceinline__ v16bf cat8(v8bf a, v8bf b) {
  return __builtin_shufflevector(a, b, 0, 1, 2, 3, 4, 5, 6, 7, 8, 9, 10, 11,
                                 12, 13, 14, 15);
}

// A-matrix 16x32 bf16 fragment from a row-major [*, 512] bf16 panel.
// lane<16: row=lane,  K = 0..7 and 16..23 ; lane>=16: row=lane-16, K=8..15,24..31
__device__ __forceinline__ v16bf load_afrag(const bf16* panel, int lane,
                                            int kt) {
  const int ksel = lane >> 4;
  const bf16* p = panel + (size_t)(lane & 15) * NN + kt * 32 + ksel * 8;
  return cat8(*(const v8bf*)p, *(const v8bf*)(p + 16));
}

// ---- prep: A -> bf16, row sums (one wave per row) --------------------------
__global__ __launch_bounds__(512) void mcrf_prep(const float* __restrict__ A,
                                                 bf16* __restrict__ Abf,
                                                 float* __restrict__ rowsum) {
  const int wave = threadIdx.x >> 5, lane = threadIdx.x & 31;
  const size_t row = (size_t)blockIdx.x * 16 + wave;  // 0 .. B*N-1
  const float* ar = A + row * NN;
  bf16* br = Abf + row * NN;
  float s = 0.f;
  for (int j = lane; j < NN; j += 32) {
    float v = ar[j];
    s += v;
    br[j] = f2bf(v);
  }
  s = wave_sum32(s);
  if (lane == 0) rowsum[row] = s;
}

// ---- loopy BP: one workgroup per batch -------------------------------------
// msg = A @ (b @ binary) via WMMA (t kept transposed in LDS, padded 16 cols).
__global__ __launch_bounds__(512) void mcrf_bp(const float* __restrict__ unary,
                                               const float* __restrict__ binary, // [B,5,5]
                                               const bf16* __restrict__ Abf,
                                               const int* __restrict__ labels,
                                               float* __restrict__ belief_out) {
  __shared__ bf16  tT[16][520];    // t transposed: [col][k], conflict-free pad
  __shared__ float msgS[512][17];
  __shared__ float logu[512][WW];
  __shared__ float bin[WW][WW];
  __shared__ int   lab[512];

  const int batch = blockIdx.x;
  const int tid = threadIdx.x;
  const int wave = tid >> 5, lane = tid & 31;

  lab[tid] = labels[batch * NN + tid];
  if (tid < 25) bin[tid / WW][tid % WW] = binary[batch * 25 + tid];
  const int myl = lab[tid];

  // log_softmax(unary) per row
  {
    const float* ur = unary + ((size_t)batch * NN + tid) * WW;
    float u[WW];
#pragma unroll
    for (int w = 0; w < WW; ++w) u[w] = ur[w];
    float mx = u[0];
#pragma unroll
    for (int w = 1; w < WW; ++w) mx = fmaxf(mx, u[w]);
    float s = 0.f;
#pragma unroll
    for (int w = 0; w < WW; ++w) s += __expf(u[w] - mx);
    const float ls = mx + __logf(s);
#pragma unroll
    for (int w = 0; w < WW; ++w) logu[tid][w] = u[w] - ls;
  }
  __syncthreads();  // bin + logu visible

  // b0 (clamped) and t0 = b0 @ bin
  {
    float b0[WW];
    if (tid < SSUP) {
#pragma unroll
      for (int w = 0; w < WW; ++w) b0[w] = (w == myl) ? 1.f : 0.f;
    } else {
#pragma unroll
      for (int w = 0; w < WW; ++w) b0[w] = __expf(logu[tid][w]);
    }
#pragma unroll
    for (int v = 0; v < WW; ++v) {
      float t = 0.f;
#pragma unroll
      for (int w = 0; w < WW; ++w) t += b0[w] * bin[w][v];
      tT[v][tid] = f2bf(t);
    }
#pragma unroll
    for (int v = WW; v < 16; ++v) tT[v][tid] = f2bf(0.f);
  }

  const int col = lane & 15;
  const int ksel = lane >> 4;
  // wave owns row tiles rt0=wave (rows wave*16..) and rt1=wave+16 (+256 rows)
  const bf16* panel0 = Abf + ((size_t)batch * NN + wave * 16) * NN;
  const bf16* panel1 = panel0 + (size_t)256 * NN;

  for (int it = 0; it < BPI; ++it) {
    __syncthreads();  // tT ready
    // --- WMMA phase: msg[512x16] = A[512x512] @ t[512x16] ---
    v8f acc0 = {0.f, 0.f, 0.f, 0.f, 0.f, 0.f, 0.f, 0.f};
    v8f acc1 = {0.f, 0.f, 0.f, 0.f, 0.f, 0.f, 0.f, 0.f};
#pragma unroll 4
    for (int kt = 0; kt < 16; ++kt) {
      const int kb = kt * 32 + ksel * 8;
      const v16bf bfr = cat8(*(const v8bf*)&tT[col][kb],
                             *(const v8bf*)&tT[col][kb + 16]);
      const v16bf af0 = load_afrag(panel0, lane, kt);
      const v16bf af1 = load_afrag(panel1, lane, kt);
      acc0 = __builtin_amdgcn_wmma_f32_16x16x32_bf16(false, af0, false, bfr,
                                                     (short)0, acc0, false,
                                                     false);
      acc1 = __builtin_amdgcn_wmma_f32_16x16x32_bf16(false, af1, false, bfr,
                                                     (short)0, acc1, false,
                                                     false);
    }
    {
      const int rbase = wave * 16 + ((lane & 16) ? 8 : 0);
#pragma unroll
      for (int m = 0; m < 8; ++m) msgS[rbase + m][col] = acc0[m];
#pragma unroll
      for (int m = 0; m < 8; ++m) msgS[rbase + 256 + m][col] = acc1[m];
    }
    __syncthreads();  // msg ready; tT reads finished

    const bool last = (it == BPI - 1);
    if (tid < SSUP) {
      if (last) {
        float* bo = belief_out + ((size_t)batch * NN + tid) * WW;
#pragma unroll
        for (int w = 0; w < WW; ++w) bo[w] = (w == myl) ? 1.f : 0.f;
      }
      // t for support rows never changes
    } else {
      float l[WW];
      float mx = -1e30f;
#pragma unroll
      for (int w = 0; w < WW; ++w) {
        l[w] = logu[tid][w] + msgS[tid][w];
        mx = fmaxf(mx, l[w]);
      }
      float s = 0.f;
#pragma unroll
      for (int w = 0; w < WW; ++w) {
        l[w] = __expf(l[w] - mx);
        s += l[w];
      }
      const float inv = 1.f / s;
      if (last) {
        float* bo = belief_out + ((size_t)batch * NN + tid) * WW;
#pragma unroll
        for (int w = 0; w < WW; ++w) bo[w] = l[w] * inv;
      } else {
#pragma unroll
        for (int v = 0; v < WW; ++v) {
          float t = 0.f;
#pragma unroll
          for (int w = 0; w < WW; ++w) t += l[w] * inv * bin[w][v];
          tT[v][tid] = f2bf(t);
        }
      }
    }
  }
}

// ---- aff update: one block per (b,i) row; K=5 dot in VALU ------------------
__global__ __launch_bounds__(512) void mcrf_aff(const float* __restrict__ A,
                                                const float* __restrict__ belief,
                                                float* __restrict__ out) {
  __shared__ float bi[WW];
  __shared__ float red[16];
  __shared__ float tot;
  const int b = blockIdx.x >> 9;
  const int i = blockIdx.x & 511;
  const int tid = threadIdx.x;
  if (tid < WW) bi[tid] = belief[((size_t)b * NN + i) * WW + tid];
  __syncthreads();
  const float* bj = belief + ((size_t)b * NN + tid) * WW;
  float d = bi[0] * bj[0] + bi[1] * bj[1] + bi[2] * bj[2] + bi[3] * bj[3] +
            bi[4] * bj[4];
  float v = A[((size_t)b * NN + i) * NN + tid] * d;
  float s = wave_sum32(v);
  if ((tid & 31) == 0) red[tid >> 5] = s;
  __syncthreads();
  if (tid == 0) {
    float r = 0.f;
#pragma unroll
    for (int k = 0; k < 16; ++k) r += red[k];
    tot = r;
  }
  __syncthreads();
  out[((size_t)b * NN + i) * NN + tid] = v / (tot + EPSF);
}

// ---- GCN: Xout = relu(row_norm(A) @ (Xin @ W) + bias), WMMA on A@Y ---------
template <int DIN, int DOUT>
__global__ __launch_bounds__(512) void mcrf_gcn(const float* __restrict__ Xin,
                                                const float* __restrict__ Wm,
                                                const float* __restrict__ bias,
                                                const bf16* __restrict__ Abf,
                                                const float* __restrict__ rowsum,
                                                float* __restrict__ Xout) {
  __shared__ bf16  YT[32][520];  // Y transposed (one 32-col half at a time)
  __shared__ float rs[512];
  const int batch = blockIdx.x;
  const int tid = threadIdx.x, wave = tid >> 5, lane = tid & 31;
  rs[tid] = rowsum[batch * NN + tid];

  const int col = lane & 15;
  const int ksel = lane >> 4;
  const bf16* panel0 = Abf + ((size_t)batch * NN + wave * 16) * NN;
  const bf16* panel1 = panel0 + (size_t)256 * NN;

  for (int h = 0; h < DOUT / 32; ++h) {
    __syncthreads();  // rs visible / previous-half WMMA reads of YT done
    // stage 1: Y[:, h*32 .. h*32+31] = Xin @ W  (row per thread)
    {
      const float* xr = Xin + ((size_t)batch * NN + tid) * DIN;
      float acc[32];
#pragma unroll
      for (int e = 0; e < 32; ++e) acc[e] = 0.f;
      for (int k0 = 0; k0 < DIN; k0 += 8) {
        float xv[8];
#pragma unroll
        for (int q = 0; q < 8; ++q) xv[q] = xr[k0 + q];
#pragma unroll
        for (int q = 0; q < 8; ++q) {
          const float* wr = Wm + (size_t)(k0 + q) * DOUT + h * 32;
#pragma unroll
          for (int e = 0; e < 32; ++e) acc[e] = fmaf(xv[q], wr[e], acc[e]);
        }
      }
#pragma unroll
      for (int e = 0; e < 32; ++e) YT[e][tid] = f2bf(acc[e]);
    }
    __syncthreads();
    // stage 2: Z = A @ Y via WMMA, 2x2 tile block per wave
    v8f acc00 = {0.f, 0.f, 0.f, 0.f, 0.f, 0.f, 0.f, 0.f};
    v8f acc01 = acc00, acc10 = acc00, acc11 = acc00;
#pragma unroll 2
    for (int kt = 0; kt < 16; ++kt) {
      const int kb = kt * 32 + ksel * 8;
      const v16bf bf0 = cat8(*(const v8bf*)&YT[col][kb],
                             *(const v8bf*)&YT[col][kb + 16]);
      const v16bf bf1 = cat8(*(const v8bf*)&YT[col + 16][kb],
                             *(const v8bf*)&YT[col + 16][kb + 16]);
      const v16bf af0 = load_afrag(panel0, lane, kt);
      const v16bf af1 = load_afrag(panel1, lane, kt);
      acc00 = __builtin_amdgcn_wmma_f32_16x16x32_bf16(false, af0, false, bf0,
                                                      (short)0, acc00, false,
                                                      false);
      acc01 = __builtin_amdgcn_wmma_f32_16x16x32_bf16(false, af0, false, bf1,
                                                      (short)0, acc01, false,
                                                      false);
      acc10 = __builtin_amdgcn_wmma_f32_16x16x32_bf16(false, af1, false, bf0,
                                                      (short)0, acc10, false,
                                                      false);
      acc11 = __builtin_amdgcn_wmma_f32_16x16x32_bf16(false, af1, false, bf1,
                                                      (short)0, acc11, false,
                                                      false);
    }
    // writeback: scale by 1/(rowsum+eps), +bias, relu
    {
      const int rbase = wave * 16 + ((lane & 16) ? 8 : 0);
      const int g0 = h * 32 + col;
      const int g1 = g0 + 16;
      const float bv0 = bias[g0];
      const float bv1 = bias[g1];
#pragma unroll
      for (int m = 0; m < 8; ++m) {
        const int r0 = rbase + m;
        const int r1 = r0 + 256;
        const float i0 = 1.f / (rs[r0] + EPSF);
        const float i1 = 1.f / (rs[r1] + EPSF);
        float* o0 = Xout + ((size_t)batch * NN + r0) * DOUT;
        float* o1 = Xout + ((size_t)batch * NN + r1) * DOUT;
        o0[g0] = fmaxf(acc00[m] * i0 + bv0, 0.f);
        o0[g1] = fmaxf(acc01[m] * i0 + bv1, 0.f);
        o1[g0] = fmaxf(acc10[m] * i1 + bv0, 0.f);
        o1[g1] = fmaxf(acc11[m] * i1 + bv1, 0.f);
      }
    }
  }
}

// ---- binary compatibility from class prototypes (tiny, VALU) ---------------
template <int DOUT>
__global__ __launch_bounds__(256) void mcrf_bc(const float* __restrict__ X,
                                               const int* __restrict__ labels,
                                               float* __restrict__ bc) {
  __shared__ float proto[WW][DOUT];
  __shared__ float cnt[WW];
  __shared__ int   labS[512];
  const int batch = blockIdx.x, tid = threadIdx.x;
  for (int n = tid; n < NN; n += 256) labS[n] = labels[batch * NN + n];
  __syncthreads();
  if (tid < WW) {
    int c = 0;
    for (int n = 0; n < NN; ++n) c += (labS[n] == tid);
    cnt[tid] = (float)c;
  }
  for (int idx = tid; idx < WW * DOUT; idx += 256) {
    const int w = idx / DOUT, d = idx % DOUT;
    float s = 0.f;
    for (int n = 0; n < NN; ++n)
      if (labS[n] == w) s += X[((size_t)batch * NN + n) * DOUT + d];
    proto[w][d] = s;
  }
  __syncthreads();
  if (tid < WW) {
    const float inv = 1.f / (cnt[tid] + EPSF);
    float n2 = 0.f;
    for (int d = 0; d < DOUT; ++d) {
      const float p = proto[tid][d] * inv;
      proto[tid][d] = p;
      n2 += p * p;
    }
    const float invn = 1.f / (sqrtf(n2) + EPSF);
    for (int d = 0; d < DOUT; ++d) proto[tid][d] *= invn;
  }
  __syncthreads();
  if (tid < WW) {
    float l[WW];
    float mx = -1e30f;
#pragma unroll
    for (int v = 0; v < WW; ++v) {
      float dd = 0.f;
      for (int d = 0; d < DOUT; ++d) dd += proto[tid][d] * proto[v][d];
      l[v] = dd;
      mx = fmaxf(mx, dd);
    }
    float s = 0.f;
#pragma unroll
    for (int v = 0; v < WW; ++v) {
      l[v] = __expf(l[v] - mx);
      s += l[v];
    }
    const float inv = 1.f / s;
#pragma unroll
    for (int v = 0; v < WW; ++v) bc[batch * 25 + tid * WW + v] = l[v] * inv;
  }
}

extern "C" void kernel_launch(void* const* d_in, const int* in_sizes, int n_in,
                              void* d_out, int out_size, void* d_ws,
                              size_t ws_size, hipStream_t stream) {
  (void)in_sizes; (void)n_in; (void)out_size; (void)ws_size;
  const float* inp_data = (const float*)d_in[0];
  const float* A        = (const float*)d_in[1];
  const float* unary    = (const float*)d_in[2];
  const float* binary0  = (const float*)d_in[3];
  const int*   labels   = (const int*)d_in[4];
  // d_in[5] all_data: unused by the reference
  const float* W1 = (const float*)d_in[6];
  const float* b1 = (const float*)d_in[7];
  const float* W2 = (const float*)d_in[8];
  const float* b2 = (const float*)d_in[9];
  // W3/b3 feed only the final (unreturned) GCN -> dead code, skipped.

  const size_t BNNs = (size_t)BB * NN * NN;  // 33,554,432
  const size_t BNWs = (size_t)BB * NN * WW;  //    327,680

  float* out  = (float*)d_out;
  float* outA = out;
  float* bel1 = out + BNNs;
  float* bel2 = bel1 + BNWs;
  float* bel3 = bel2 + BNWs;
  float* aff1 = bel3 + BNWs;
  float* aff2 = aff1 + BNNs;
  float* aff3 = aff2 + BNNs;

  uint8_t* w = (uint8_t*)d_ws;
  bf16*  Abf    = (bf16*)w;  w += BNNs * sizeof(bf16);          // 64 MiB
  float* rowsum = (float*)w; w += (size_t)BB * NN * 4;          // 256 KiB
  float* X1     = (float*)w; w += (size_t)BB * NN * 64 * 4;     // 16 MiB
  float* X2     = (float*)w; w += (size_t)BB * NN * 32 * 4;     //  8 MiB
  float* bc1    = (float*)w; w += (size_t)BB * 25 * 4;
  float* bc2    = (float*)w;

  // output A = input affinity (copied verbatim)
  hipMemcpyAsync(outA, A, BNNs * sizeof(float), hipMemcpyDeviceToDevice,
                 stream);

  mcrf_prep<<<(BB * NN) / 16, 512, 0, stream>>>(A, Abf, rowsum);

  // round 1
  mcrf_bp<<<BB, 512, 0, stream>>>(unary, binary0, Abf, labels, bel1);
  mcrf_aff<<<BB * NN, 512, 0, stream>>>(A, bel1, aff1);
  mcrf_gcn<128, 64><<<BB, 512, 0, stream>>>(inp_data, W1, b1, Abf, rowsum, X1);
  mcrf_bc<64><<<BB, 256, 0, stream>>>(X1, labels, bc1);

  // round 2
  mcrf_bp<<<BB, 512, 0, stream>>>(unary, bc1, Abf, labels, bel2);
  mcrf_aff<<<BB * NN, 512, 0, stream>>>(A, bel2, aff2);
  mcrf_gcn<64, 32><<<BB, 512, 0, stream>>>(X1, W2, b2, Abf, rowsum, X2);
  mcrf_bc<32><<<BB, 256, 0, stream>>>(X2, labels, bc2);

  // round 3 (GCN with W3 is never returned -> skipped)
  mcrf_bp<<<BB, 512, 0, stream>>>(unary, bc2, Abf, labels, bel3);
  mcrf_aff<<<BB * NN, 512, 0, stream>>>(A, bel3, aff3);
}
